// MeshConvPoint_23441931502099
// MI455X (gfx1250) — matrix operand order repaired and checked by reading, not compiled, not run on hardware
//
#include <hip/hip_runtime.h>
#include <math.h>

#define B_     4
#define CIN    64
#define V_     50000
#define COUT   128
#define KT     7
#define KK     448        // CIN * KT
#define KSTEPS 14         // KK / 32
#define MT     64         // points per block
#define APITCH 928        // LDS A-row pitch in bytes (448*2 + 32 pad, 16B aligned)

typedef __bf16        v16bf __attribute__((ext_vector_type(16)));
typedef float         v8f   __attribute__((ext_vector_type(8)));
typedef unsigned int  u32x4 __attribute__((ext_vector_type(4)));

union FragU {
  v16bf v;
  u32x4 q[2];
};

__device__ __forceinline__ unsigned short f2bf(float f) {
  unsigned int u = __float_as_uint(f);
  u += 0x7fffu + ((u >> 16) & 1u);   // round-to-nearest-even
  return (unsigned short)(u >> 16);
}
__device__ __forceinline__ float bf2f(unsigned short h) {
  return __uint_as_float(((unsigned int)h) << 16);
}

// ---------- kernel 1: W f32 -> bf16, flat [COUT][KK] (kk = c*7 + k) ----------
__global__ __launch_bounds__(256)
void wconv_bf16(const float* __restrict__ W, unsigned short* __restrict__ Wb) {
  int i = blockIdx.x * 256 + threadIdx.x;
  if (i < COUT * KK) Wb[i] = f2bf(W[i]);
}

// ---------- kernel 2: x [B,C,V] f32 -> xT [B,V,C] bf16 (LDS tiled) ----------
__global__ __launch_bounds__(256)
void xpose_bf16(const float* __restrict__ x, unsigned short* __restrict__ xT) {
  const int NT = (V_ + 63) / 64;
  const int bb = blockIdx.x / NT;
  const int vt = blockIdx.x % NT;
  const int v0 = vt * 64;
  __shared__ float sm[64][65];
  const int t = threadIdx.x;
  #pragma unroll
  for (int i = 0; i < 16; ++i) {
    int q = i * 256 + t;
    int c = q >> 6, vv = q & 63;
    int v = v0 + vv;
    sm[vv][c] = (v < V_) ? x[((size_t)bb * CIN + c) * V_ + v] : 0.f;
  }
  __syncthreads();
  #pragma unroll
  for (int i = 0; i < 16; ++i) {
    int q = i * 256 + t;
    int vv = q >> 6, c = q & 63;
    if (v0 + vv < V_)
      xT[((size_t)bb * V_ + v0 + vv) * CIN + c] = f2bf(sm[vv][c]);
  }
}

// ---------- kernel 3: gather + symmetric ops + WMMA GEMM ----------
__global__ __launch_bounds__(256)
void mesh_conv_main(const unsigned short* __restrict__ xT,
                    const unsigned short* __restrict__ Wb,
                    const int* __restrict__ Gi,
                    const float* __restrict__ bias,
                    float* __restrict__ out) {
  __shared__ __attribute__((aligned(16))) unsigned char smA[MT * APITCH];
  __shared__ int sIdx[MT][4];
  __shared__ int sB[MT];

  const int  t  = threadIdx.x;
  const long pb = (long)blockIdx.x * MT;

  // stage neighbor indices + per-point batch id (256 threads == 64 pts * 4 nbrs)
  {
    int p = t >> 2, j = t & 3;
    long P  = pb + p;
    int  bb = (int)(P / V_);
    int  vv = (int)(P - (long)bb * V_);
    sIdx[p][j] = Gi[((long)bb * V_ + vv) * 4 + j];
    if (j == 0) sB[p] = bb;
  }
  __syncthreads();

  // Phase 1: build 64x448 bf16 A-tile of symmetric features in LDS
  #pragma unroll 4
  for (int i = 0; i < 16; ++i) {
    int q = i * 256 + t;          // 0..4095 : 64 points x 64 channels
    int p = q >> 6;
    int c = q & 63;
    const unsigned short* xb = xT + (size_t)sB[p] * V_ * CIN + c;
    float f0 = bf2f(xb[(size_t)sIdx[p][0] * CIN]);
    float f1 = bf2f(xb[(size_t)sIdx[p][1] * CIN]);
    float f2 = bf2f(xb[(size_t)sIdx[p][2] * CIN]);
    float f3 = bf2f(xb[(size_t)sIdx[p][3] * CIN]);
    float g1 = f1 + f2 + f3;
    float g2 = f1 * f2 * f3;
    float g3 = f1 * f2 + f1 * f3 + f2 * f3;
    float g4 = f1 * f1 + f2 * f2 + f3 * f3;
    float g5 = fabsf(f1 - f2) + fabsf(f1 - f3) + fabsf(f2 - f3);
    float g6 = f1 * f1 * f1 + f2 * f2 * f2 + f3 * f3 * f3;
    unsigned short* arow = (unsigned short*)(smA + p * APITCH) + c * KT;
    arow[0] = f2bf(f0); arow[1] = f2bf(g1); arow[2] = f2bf(g2);
    arow[3] = f2bf(g3); arow[4] = f2bf(g4); arow[5] = f2bf(g5);
    arow[6] = f2bf(g6);
  }
  __syncthreads();

  // Phase 2: [64 x 448] x [448 x 128] with v_wmma_f32_16x16x32_bf16
  const int lane    = t & 31;
  const int wave    = t >> 5;
  const int hi      = lane >> 4;          // half-wave select
  const int l15     = lane & 15;
  const int o       = wave * 16 + l15;    // output channel for this lane

  v8f acc[4] = {};                         // 4 row tiles x 8 VGPRs f32

  // B fragment base: lane holds K = 8*hi+0..7 and 16+8*hi+0..7 of column o
  const unsigned char* wrow =
      (const unsigned char*)Wb + ((size_t)o * KK + hi * 8) * 2;
  const unsigned char* abase = smA + l15 * APITCH + hi * 16;

  for (int ks = 0; ks < KSTEPS; ++ks) {
    // issue ALL loads for this k-step first (independent regs -> partial waits,
    // LDS latency overlapped with matrix-pipe execution)
    FragU bf_;
    const unsigned char* wp = wrow + ks * 64;
    bf_.q[0] = *(const u32x4*)(wp);
    bf_.q[1] = *(const u32x4*)(wp + 32);

    FragU af[4];
    #pragma unroll
    for (int r = 0; r < 4; ++r) {
      const unsigned char* ap = abase + r * (16 * APITCH) + ks * 64;
      af[r].q[0] = *(const u32x4*)(ap);        // K = base+0..7
      af[r].q[1] = *(const u32x4*)(ap + 32);   // K = base+16..23
    }

    #pragma unroll
    for (int r = 0; r < 4; ++r) {
      acc[r] = __builtin_amdgcn_wmma_f32_16x16x32_bf16(
          false, af[r].v, false, bf_.v, (short)0, acc[r], false, false);
    }
  }

  // Epilogue: D layout -> out[b][o][v] + bias
  const float bv = bias[o];
  #pragma unroll
  for (int r = 0; r < 4; ++r) {
    #pragma unroll
    for (int j = 0; j < 8; ++j) {
      int  p  = r * 16 + hi * 8 + j;        // point row M
      int  bb = sB[p];
      long P  = pb + p;
      int  vv = (int)(P - (long)bb * V_);
      out[((size_t)bb * COUT + o) * V_ + vv] = acc[r][j] + bv;
    }
  }
}

extern "C" void kernel_launch(void* const* d_in, const int* in_sizes, int n_in,
                              void* d_out, int out_size, void* d_ws, size_t ws_size,
                              hipStream_t stream) {
  const float* x  = (const float*)d_in[0];
  const int*   Gi = (const int*)d_in[1];
  const float* W  = (const float*)d_in[2];
  const float* bs = (const float*)d_in[3];
  float*       out = (float*)d_out;

  // workspace layout: [0,128K) Wb bf16 ; [128K, 128K+25.6MB) xT bf16
  unsigned short* Wb = (unsigned short*)d_ws;
  unsigned short* xT = (unsigned short*)((char*)d_ws + (128 << 10));

  wconv_bf16<<<(COUT * KK + 255) / 256, 256, 0, stream>>>(W, Wb);

  const int NT = (V_ + 63) / 64;
  xpose_bf16<<<B_ * NT, 256, 0, stream>>>(x, xT);

  mesh_conv_main<<<(B_ * V_) / MT, 256, 0, stream>>>(xT, Wb, Gi, bs, out);
}
